// PrototypicalNetworksKNN_9878424781139
// MI455X (gfx1250) — matrix operand order repaired and checked by reading, not compiled, not run on hardware
//
#include <hip/hip_runtime.h>
#include <cstddef>

// ---------------------------------------------------------------------------
// PrototypicalNetworks + kNN scoring for MI455X (gfx1250), wave32 + WMMA bf16.
//   1) fp32 -> bf16 conversion of X_s, X_q; W converted+transposed to [DF][DIN]
//   2) Z = X @ W via v_wmma_f32_16x16x32_bf16 (ping-pong pipelined K loop)
//   3) row norms ||z||^2 (wave reductions)
//   4) class prototypes (per-class scan, no atomics) + proto norms
//   5) proto distances [NQ x 5] (VALU, tiny)
//   6) streaming kNN: per wave, 16 queries x 64 support per pass via WMMA,
//      squared-distance top-10 kept in registers, label histogram -> ws
//   7) scores = -(proto_dist / (hist^3 + 1))
// ---------------------------------------------------------------------------

typedef __attribute__((ext_vector_type(16))) __bf16 v16bf;
typedef __attribute__((ext_vector_type(8)))  __bf16 v8bf;
typedef __attribute__((ext_vector_type(8)))  float  v8f;

#define NSUP 5000
#define NQRY 10000
#define DIN  2048
#define DFEA 1024
#define KNN  10
#define NWAY 5

// A fragment (16x32 bf16, M x K): lane 0-15 row M=lane; K groups of 8 with a
// 16-element stride between halves => two 16B loads at laneHalf*8, +16.
__device__ __forceinline__ v16bf load_fragA(const __bf16* row, int k0, int laneHalf) {
    const v8bf lo = *(const v8bf*)(row + k0 + laneHalf * 8);
    const v8bf hi = *(const v8bf*)(row + k0 + laneHalf * 8 + 16);
    v16bf f;
#pragma unroll
    for (int i = 0; i < 8; ++i) { f[i] = lo[i]; f[8 + i] = hi[i]; }
    return f;
}

// B fragment (32x16 bf16, K x N): col N=lane%16; same K order as A fragment
// => two 16B loads at laneHalf*8, +16.
__device__ __forceinline__ v16bf load_fragB(const __bf16* row, int k0, int laneHalf) {
    const v8bf lo = *(const v8bf*)(row + k0 + laneHalf * 8);
    const v8bf hi = *(const v8bf*)(row + k0 + laneHalf * 8 + 16);
    v16bf f;
#pragma unroll
    for (int i = 0; i < 8; ++i) { f[i] = lo[i]; f[8 + i] = hi[i]; }
    return f;
}

__device__ __forceinline__ v8f wmma_bf16(v16bf a, v16bf b, v8f c) {
    return __builtin_amdgcn_wmma_f32_16x16x32_bf16(false, a, false, b, (short)0, c,
                                                   false, false);
}

// ------------------------------- conversions --------------------------------
__global__ __launch_bounds__(256) void cvt_f32_bf16(const float* __restrict__ x,
                                                    __bf16* __restrict__ y, size_t n) {
    size_t i = (size_t)blockIdx.x * 256 + threadIdx.x;
    size_t stride = (size_t)gridDim.x * 256;
    for (; i < n; i += stride) y[i] = (__bf16)x[i];
}

__global__ __launch_bounds__(256) void cvt_transpose_W(const float* __restrict__ W,
                                                       __bf16* __restrict__ Wt) {
    int idx = blockIdx.x * 256 + threadIdx.x;     // over DIN*DFEA, coalesced read
    if (idx < DIN * DFEA) {
        int k = idx / DFEA, n = idx % DFEA;
        Wt[(size_t)n * DIN + k] = (__bf16)W[idx];
    }
}

// ------------------------------- GEMM (WMMA) --------------------------------
// C[M x N] (bf16 out) = A[M x Kd] @ Bt[N x Kd]^T. Block = 4 waves, each wave
// owns a 16(M) x 64(N) tile. K loop ping-pong pipelined (2 k-steps per
// iteration, alternating fragment buffers -> no register rotation moves).
// Kd must be a multiple of 64 (2048 and 1024 here).
__global__ __launch_bounds__(128) void gemm_bf16(const __bf16* __restrict__ A,
                                                 const __bf16* __restrict__ Bt,
                                                 __bf16* __restrict__ C,
                                                 int M, int N, int Kd) {
    const int lane = threadIdx.x & 31;
    const int wave = threadIdx.x >> 5;
    const int laneHalf = lane >> 4;
    const int lane15 = lane & 15;
    const int tileN = blockIdx.x * 64;
    const int tileM = blockIdx.y * 64 + wave * 16;

    int mRow = tileM + lane15; if (mRow >= M) mRow = M - 1;   // clamp for loads
    const __bf16* aRow = A + (size_t)mRow * Kd;
    const __bf16* bRow[4];
#pragma unroll
    for (int j = 0; j < 4; ++j)
        bRow[j] = Bt + (size_t)(tileN + j * 16 + lane15) * Kd;

    v8f acc[4];
#pragma unroll
    for (int j = 0; j < 4; ++j) acc[j] = (v8f){};

    // ping buffer: k = 0
    v16bf aA = load_fragA(aRow, 0, laneHalf);
    v16bf bA[4];
#pragma unroll
    for (int j = 0; j < 4; ++j) bA[j] = load_fragB(bRow[j], 0, laneHalf);

    v16bf aB;
    v16bf bB[4];
    for (int k0 = 0; k0 < Kd - 64; k0 += 64) {
        // pong loads (k0+32), then consume ping
        aB = load_fragA(aRow, k0 + 32, laneHalf);
#pragma unroll
        for (int j = 0; j < 4; ++j) bB[j] = load_fragB(bRow[j], k0 + 32, laneHalf);
#pragma unroll
        for (int j = 0; j < 4; ++j) acc[j] = wmma_bf16(aA, bA[j], acc[j]);
        // ping loads (k0+64), then consume pong
        aA = load_fragA(aRow, k0 + 64, laneHalf);
#pragma unroll
        for (int j = 0; j < 4; ++j) bA[j] = load_fragB(bRow[j], k0 + 64, laneHalf);
#pragma unroll
        for (int j = 0; j < 4; ++j) acc[j] = wmma_bf16(aB, bB[j], acc[j]);
    }
    // epilogue: last pair (Kd-64 in ping, load Kd-32 into pong)
    aB = load_fragA(aRow, Kd - 32, laneHalf);
#pragma unroll
    for (int j = 0; j < 4; ++j) bB[j] = load_fragB(bRow[j], Kd - 32, laneHalf);
#pragma unroll
    for (int j = 0; j < 4; ++j) acc[j] = wmma_bf16(aA, bA[j], acc[j]);
#pragma unroll
    for (int j = 0; j < 4; ++j) acc[j] = wmma_bf16(aB, bB[j], acc[j]);

#pragma unroll
    for (int j = 0; j < 4; ++j) {
#pragma unroll
        for (int r = 0; r < 8; ++r) {
            int m = tileM + laneHalf * 8 + r;
            int n = tileN + j * 16 + lane15;
            if (m < M) C[(size_t)m * N + n] = (__bf16)acc[j][r];
        }
    }
}

// ------------------------------- row norms ----------------------------------
__global__ __launch_bounds__(256) void row_norms(const __bf16* __restrict__ Z,
                                                 float* __restrict__ norms, int R) {
    int row = blockIdx.x * 8 + (threadIdx.x >> 5);
    int lane = threadIdx.x & 31;
    if (row >= R) return;
    const __bf16* z = Z + (size_t)row * DFEA;
    float s = 0.f;
    for (int i = lane; i < DFEA; i += 32) { float v = (float)z[i]; s += v * v; }
#pragma unroll
    for (int off = 16; off > 0; off >>= 1) s += __shfl_down(s, off, 32);
    if (lane == 0) norms[row] = s;
}

// ------------------------------ prototypes ----------------------------------
__global__ __launch_bounds__(256) void proto_accum(const __bf16* __restrict__ Zs,
                                                   const int* __restrict__ labels,
                                                   float* __restrict__ protoSum) {
    int c = blockIdx.x;
    int col = blockIdx.y * 256 + threadIdx.x;
    float acc = 0.f;
    for (int s = 0; s < NSUP; ++s) {
        if (labels[s] == c) acc += (float)Zs[(size_t)s * DFEA + col];
    }
    protoSum[c * DFEA + col] = acc;
}

__global__ __launch_bounds__(256) void proto_final(const float* __restrict__ protoSum,
                                                   const int* __restrict__ labels,
                                                   __bf16* __restrict__ protoBf,
                                                   float* __restrict__ protoNorm) {
    int c = blockIdx.x;
    __shared__ float red[256];
    int cnt = 0;
    for (int s = (int)threadIdx.x; s < NSUP; s += 256) cnt += (labels[s] == c) ? 1 : 0;
    red[threadIdx.x] = (float)cnt;
    __syncthreads();
    for (int off = 128; off > 0; off >>= 1) {
        if ((int)threadIdx.x < off) red[threadIdx.x] += red[threadIdx.x + off];
        __syncthreads();
    }
    float count = red[0] > 0.f ? red[0] : 1.f;
    __syncthreads();
    float nrm = 0.f;
    for (int i = (int)threadIdx.x; i < DFEA; i += 256) {
        float p = protoSum[c * DFEA + i] / count;
        protoBf[c * DFEA + i] = (__bf16)p;
        nrm += p * p;
    }
    red[threadIdx.x] = nrm;
    __syncthreads();
    for (int off = 128; off > 0; off >>= 1) {
        if ((int)threadIdx.x < off) red[threadIdx.x] += red[threadIdx.x + off];
        __syncthreads();
    }
    if (threadIdx.x == 0) protoNorm[c] = red[0];
}

// --------------------------- prototype distances ----------------------------
__global__ __launch_bounds__(256) void proto_dist(const __bf16* __restrict__ Zq,
                                                  const __bf16* __restrict__ protoBf,
                                                  const float* __restrict__ nq,
                                                  const float* __restrict__ protoNorm,
                                                  float* __restrict__ protod) {
    int q = blockIdx.x * 8 + (threadIdx.x >> 5);
    int lane = threadIdx.x & 31;
    if (q >= NQRY) return;
    const __bf16* zq = Zq + (size_t)q * DFEA;
    float dot[NWAY] = {0.f, 0.f, 0.f, 0.f, 0.f};
    for (int i = lane; i < DFEA; i += 32) {
        float v = (float)zq[i];
#pragma unroll
        for (int c = 0; c < NWAY; ++c) dot[c] += v * (float)protoBf[c * DFEA + i];
    }
#pragma unroll
    for (int c = 0; c < NWAY; ++c)
#pragma unroll
        for (int off = 16; off > 0; off >>= 1) dot[c] += __shfl_down(dot[c], off, 32);
    if (lane == 0) {
        float a2 = nq[q];
#pragma unroll
        for (int c = 0; c < NWAY; ++c) {
            float sq = a2 + protoNorm[c] - 2.f * dot[c];
            protod[q * NWAY + c] = sqrtf(fmaxf(sq, 1e-12f));
        }
    }
}

// -------------------- streaming kNN (WMMA Gram + top-10) --------------------
// One wave per 16-query tile: loop support in 64-wide blocks (4 WMMA tiles
// share one A fragment), ping-pong pipelined K loop, squared distances via a
// 16x64 LDS tile, register top-10 per query row, then class histogram.
__global__ __launch_bounds__(32) void knn_topk(const __bf16* __restrict__ Zq,
                                               const __bf16* __restrict__ Zs,
                                               const float* __restrict__ nq,
                                               const float* __restrict__ ns,
                                               const int* __restrict__ labels,
                                               float* __restrict__ hist) {
    __shared__ float tile[16 * 64];
    const int lane = threadIdx.x;
    const int laneHalf = lane >> 4;
    const int lane15 = lane & 15;
    const int qBase = blockIdx.x * 16;            // NQRY % 16 == 0

    const __bf16* aRow = Zq + (size_t)(qBase + lane15) * DFEA;

    float qn[8];
#pragma unroll
    for (int r = 0; r < 8; ++r) qn[r] = nq[qBase + laneHalf * 8 + r];

    const float INIT = 3.4e38f;
    float bd[KNN]; int bi[KNN];
#pragma unroll
    for (int i = 0; i < KNN; ++i) { bd[i] = INIT; bi[i] = 0; }

    for (int sBase = 0; sBase < NSUP; sBase += 64) {
        const __bf16* bRow[4];
        int sIdx[4];
#pragma unroll
        for (int j = 0; j < 4; ++j) {
            sIdx[j] = sBase + j * 16 + lane15;
            int sr = sIdx[j] < NSUP ? sIdx[j] : NSUP - 1;
            bRow[j] = Zs + (size_t)sr * DFEA;
        }
        if (sBase + 64 < NSUP) {                  // global_prefetch_b8 next block
#pragma unroll
            for (int j = 0; j < 4; ++j)
                __builtin_prefetch(Zs + (size_t)(sBase + 64 + j * 16 + lane15) * DFEA, 0, 3);
        }

        v8f acc[4];
#pragma unroll
        for (int j = 0; j < 4; ++j) acc[j] = (v8f){};

        v16bf aA = load_fragA(aRow, 0, laneHalf);
        v16bf bA[4];
#pragma unroll
        for (int j = 0; j < 4; ++j) bA[j] = load_fragB(bRow[j], 0, laneHalf);

        v16bf aB;
        v16bf bB[4];
        for (int k0 = 0; k0 < DFEA - 64; k0 += 64) {
            aB = load_fragA(aRow, k0 + 32, laneHalf);
#pragma unroll
            for (int j = 0; j < 4; ++j) bB[j] = load_fragB(bRow[j], k0 + 32, laneHalf);
#pragma unroll
            for (int j = 0; j < 4; ++j) acc[j] = wmma_bf16(aA, bA[j], acc[j]);
            aA = load_fragA(aRow, k0 + 64, laneHalf);
#pragma unroll
            for (int j = 0; j < 4; ++j) bA[j] = load_fragB(bRow[j], k0 + 64, laneHalf);
#pragma unroll
            for (int j = 0; j < 4; ++j) acc[j] = wmma_bf16(aB, bB[j], acc[j]);
        }
        aB = load_fragA(aRow, DFEA - 32, laneHalf);
#pragma unroll
        for (int j = 0; j < 4; ++j) bB[j] = load_fragB(bRow[j], DFEA - 32, laneHalf);
#pragma unroll
        for (int j = 0; j < 4; ++j) acc[j] = wmma_bf16(aA, bA[j], acc[j]);
#pragma unroll
        for (int j = 0; j < 4; ++j) acc[j] = wmma_bf16(aB, bB[j], acc[j]);

        // D element (m = laneHalf*8 + r, n = j*16 + lane15) -> squared distance
#pragma unroll
        for (int j = 0; j < 4; ++j) {
            float sNorm = (sIdx[j] < NSUP) ? ns[sIdx[j]] : 0.f;
#pragma unroll
            for (int r = 0; r < 8; ++r) {
                int m = laneHalf * 8 + r;
                float d2 = (sIdx[j] < NSUP) ? (qn[r] + sNorm - 2.f * acc[j][r]) : INIT;
                tile[m * 64 + j * 16 + lane15] = d2;
            }
        }
        __syncthreads();
        if (lane < 16) {
            for (int j = 0; j < 64; ++j) {        // scan 64 candidates
                float v = tile[lane * 64 + j];
                if (v < bd[KNN - 1]) {
                    bd[KNN - 1] = v; bi[KNN - 1] = sBase + j;
#pragma unroll
                    for (int i = KNN - 1; i > 0; --i) {  // static-indexed insert
                        if (bd[i] < bd[i - 1]) {
                            float tv = bd[i]; bd[i] = bd[i - 1]; bd[i - 1] = tv;
                            int   ti = bi[i]; bi[i] = bi[i - 1]; bi[i - 1] = ti;
                        }
                    }
                }
            }
        }
        __syncthreads();
    }

    if (lane < 16) {
        int q = qBase + lane;
        float h[NWAY] = {0.f, 0.f, 0.f, 0.f, 0.f};
#pragma unroll
        for (int i = 0; i < KNN; ++i) {
            int lb = labels[bi[i]];
#pragma unroll
            for (int c = 0; c < NWAY; ++c) h[c] += (lb == c) ? 1.f : 0.f;
        }
#pragma unroll
        for (int c = 0; c < NWAY; ++c) hist[(size_t)q * NWAY + c] = h[c];
    }
}

// ------------------------------ final scores --------------------------------
__global__ __launch_bounds__(256) void final_scores(const float* __restrict__ protod,
                                                    const float* __restrict__ hist,
                                                    float* __restrict__ out, int n) {
    int i = blockIdx.x * 256 + threadIdx.x;
    if (i < n) {
        float h = hist[i];
        out[i] = -(protod[i] / (h * h * h + 1.0f));
    }
}

// ------------------------------- launcher -----------------------------------
extern "C" void kernel_launch(void* const* d_in, const int* in_sizes, int n_in,
                              void* d_out, int out_size, void* d_ws, size_t ws_size,
                              hipStream_t stream) {
    const float* Xs = (const float*)d_in[0];   // [5000 x 2048]
    const float* Xq = (const float*)d_in[1];   // [10000 x 2048]
    const float* W  = (const float*)d_in[2];   // [2048 x 1024]
    const int* labels = (const int*)d_in[3];   // [5000]
    float* out = (float*)d_out;                // [10000 x 5]
    (void)in_sizes; (void)n_in; (void)out_size; (void)ws_size;

    char* ws = (char*)d_ws;
    size_t off = 0;
    auto alloc = [&](size_t bytes) -> void* {
        void* p = (void*)(ws + off);
        off = (off + bytes + 255) & ~(size_t)255;
        return p;
    };
    __bf16* XsB  = (__bf16*)alloc((size_t)NSUP * DIN * 2);
    __bf16* XqB  = (__bf16*)alloc((size_t)NQRY * DIN * 2);
    __bf16* WtB  = (__bf16*)alloc((size_t)DFEA * DIN * 2);
    __bf16* ZsB  = (__bf16*)alloc((size_t)NSUP * DFEA * 2);
    __bf16* ZqB  = (__bf16*)alloc((size_t)NQRY * DFEA * 2);
    __bf16* PrB  = (__bf16*)alloc((size_t)NWAY * DFEA * 2);
    float*  nsArr  = (float*)alloc((size_t)NSUP * 4);
    float*  nqArr  = (float*)alloc((size_t)NQRY * 4);
    float*  pSum   = (float*)alloc((size_t)NWAY * DFEA * 4);
    float*  pNorm  = (float*)alloc((size_t)NWAY * 4);
    float*  protod = (float*)alloc((size_t)NQRY * NWAY * 4);
    float*  histB  = (float*)alloc((size_t)NQRY * NWAY * 4);

    cvt_f32_bf16<<<8192, 256, 0, stream>>>(Xs, XsB, (size_t)NSUP * DIN);
    cvt_f32_bf16<<<8192, 256, 0, stream>>>(Xq, XqB, (size_t)NQRY * DIN);
    cvt_transpose_W<<<(DIN * DFEA + 255) / 256, 256, 0, stream>>>(W, WtB);

    {
        dim3 g(DFEA / 64, (NSUP + 63) / 64);
        gemm_bf16<<<g, 128, 0, stream>>>(XsB, WtB, ZsB, NSUP, DFEA, DIN);
    }
    {
        dim3 g(DFEA / 64, (NQRY + 63) / 64);
        gemm_bf16<<<g, 128, 0, stream>>>(XqB, WtB, ZqB, NQRY, DFEA, DIN);
    }

    row_norms<<<(NSUP + 7) / 8, 256, 0, stream>>>(ZsB, nsArr, NSUP);
    row_norms<<<(NQRY + 7) / 8, 256, 0, stream>>>(ZqB, nqArr, NQRY);

    {
        dim3 g(NWAY, DFEA / 256);
        proto_accum<<<g, 256, 0, stream>>>(ZsB, labels, pSum);
    }
    proto_final<<<NWAY, 256, 0, stream>>>(pSum, labels, PrB, pNorm);

    proto_dist<<<(NQRY + 7) / 8, 256, 0, stream>>>(ZqB, PrB, nqArr, pNorm, protod);

    knn_topk<<<NQRY / 16, 32, 0, stream>>>(ZqB, ZsB, nqArr, nsArr, labels, histB);

    final_scores<<<(NQRY * NWAY + 255) / 256, 256, 0, stream>>>(protod, histB, out,
                                                                NQRY * NWAY);
}
